// WaveGlow_41077067218947
// MI455X (gfx1250) — matrix operand order repaired and checked by reading, not compiled, not run on hardware
//
#include <hip/hip_runtime.h>
#include <math.h>

#define T_LEN  2048
#define NBATCH 8
#define NC     128
#define NTILE  64
#define SXW    80      // x tile columns incl. max halo (64 + 2*8)
#define SXT    200     // transposed x tile col stride (u16): 400B, 16B-aligned, bank-stride 100
#define SAT    136     // transposed act tile col stride (u16): 272B, 16B-aligned, bank-stride 68

typedef __attribute__((ext_vector_type(16))) __bf16 v16bf;
typedef __attribute__((ext_vector_type(8)))  float  v8f;

__device__ inline unsigned short f2bf(float f) {
    union { float f; unsigned u; } v; v.f = f;
    unsigned u = v.u;
    unsigned r = u + 0x7FFFu + ((u >> 16) & 1u);
    if ((u & 0x7F800000u) == 0x7F800000u) r = u;   // inf/nan passthrough
    return (unsigned short)(r >> 16);
}
__device__ inline float bf2f(unsigned short s) {
    union { unsigned u; float f; } o; o.u = ((unsigned)s) << 16; return o.f;
}

// ---------------------------------------------------------------------------
// prep: context f32 -> bf16, forecast (B, T*8) -> x layout (B, 8, T) f32
// ---------------------------------------------------------------------------
__global__ void __launch_bounds__(256) k_prep(const float* __restrict__ ctx_f,
                                              unsigned short* __restrict__ ctx_bf,
                                              const float* __restrict__ fc,
                                              float* __restrict__ xw) {
    int idx = blockIdx.x * blockDim.x + threadIdx.x;
    if (idx < NBATCH * 64 * T_LEN) ctx_bf[idx] = f2bf(ctx_f[idx]);
    if (idx < NBATCH * 8 * T_LEN) {
        int b  = idx / (8 * T_LEN);
        int g  = (idx / T_LEN) & 7;
        int tt = idx % T_LEN;
        xw[idx] = fc[b * 16384 + tt * 8 + g];
    }
}

// ---------------------------------------------------------------------------
// slogdet of 8 small matrices (8x8 for flows 0-3, 6x6 for 4-7) via LU
// ---------------------------------------------------------------------------
__global__ void k_logdet(const float* w0, const float* w1, const float* w2, const float* w3,
                         const float* w4, const float* w5, const float* w6, const float* w7,
                         float* __restrict__ out) {
    int k = threadIdx.x;
    if (k >= 8) return;
    const float* ws[8] = {w0, w1, w2, w3, w4, w5, w6, w7};
    int n = (k < 4) ? 8 : 6;
    float m[64];
    for (int i = 0; i < n; ++i)
        for (int j = 0; j < n; ++j) m[i * 8 + j] = ws[k][i * n + j];
    float ld = 0.0f;
    for (int i = 0; i < n; ++i) {
        int piv = i; float best = fabsf(m[i * 8 + i]);
        for (int r = i + 1; r < n; ++r) {
            float a = fabsf(m[r * 8 + i]);
            if (a > best) { best = a; piv = r; }
        }
        if (piv != i)
            for (int c = 0; c < n; ++c) {
                float t = m[i * 8 + c]; m[i * 8 + c] = m[piv * 8 + c]; m[piv * 8 + c] = t;
            }
        float p = m[i * 8 + i];
        ld += logf(fabsf(p));
        for (int r = i + 1; r < n; ++r) {
            float f = m[r * 8 + i] / p;
            for (int c = i; c < n; ++c) m[r * 8 + c] -= f * m[i * 8 + c];
        }
    }
    out[k] = 16384.0f * ld;   // b * t = 8 * 2048
}

// ---------------------------------------------------------------------------
// early output at flow 4: emit first 2 channels to z, shift remaining 6 down
// ---------------------------------------------------------------------------
__global__ void __launch_bounds__(256) k_early(float* __restrict__ xw, float* __restrict__ z) {
    int idx = blockIdx.x * blockDim.x + threadIdx.x;
    if (idx >= NBATCH * T_LEN) return;
    int b = idx / T_LEN, tt = idx % T_LEN;
    z[(b * 8 + 0) * T_LEN + tt] = xw[(b * 8 + 0) * T_LEN + tt];
    z[(b * 8 + 1) * T_LEN + tt] = xw[(b * 8 + 1) * T_LEN + tt];
    float tmp[6];
    for (int r = 0; r < 6; ++r) tmp[r] = xw[(b * 8 + 2 + r) * T_LEN + tt];
    for (int r = 0; r < 6; ++r) xw[(b * 8 + r) * T_LEN + tt] = tmp[r];
}

// ---------------------------------------------------------------------------
// invertible 1x1 conv: per-time-column x' = ic_w @ x ; also emits x0 as bf16
// ---------------------------------------------------------------------------
__global__ void __launch_bounds__(256) k_invconv(const float* __restrict__ icw,
                                                 float* __restrict__ xw,
                                                 unsigned short* __restrict__ x0bf,
                                                 int nrem, int n_half) {
    int idx = blockIdx.x * blockDim.x + threadIdx.x;
    if (idx >= NBATCH * T_LEN) return;
    int b = idx / T_LEN, tt = idx % T_LEN;
    float v[8], y[8];
    for (int i = 0; i < nrem; ++i) v[i] = xw[(b * 8 + i) * T_LEN + tt];
    for (int o = 0; o < nrem; ++o) {
        float s = 0.0f;
        for (int i = 0; i < nrem; ++i) s += icw[o * nrem + i] * v[i];
        y[o] = s;
    }
    for (int o = 0; o < nrem; ++o) xw[(b * 8 + o) * T_LEN + tt] = y[o];
    for (int r = 0; r < n_half; ++r) x0bf[(b * 4 + r) * T_LEN + tt] = f2bf(y[r]);
}

// ---------------------------------------------------------------------------
// start 1x1 conv (K = n_half <= 4): h = W x0 + b, also zeroes skip accumulator
// ---------------------------------------------------------------------------
__global__ void __launch_bounds__(256) k_start(const float* __restrict__ sw_,
                                               const float* __restrict__ sb,
                                               const unsigned short* __restrict__ x0bf,
                                               unsigned short* __restrict__ h,
                                               float* __restrict__ outacc, int n_half) {
    int idx = blockIdx.x * blockDim.x + threadIdx.x;
    if (idx >= NBATCH * NC * T_LEN) return;
    int b  = idx / (NC * T_LEN);
    int c  = (idx / T_LEN) % NC;
    int tt = idx % T_LEN;
    float s = sb[c];
    for (int i = 0; i < n_half; ++i)
        s += sw_[c * n_half + i] * bf2f(x0bf[(b * 4 + i) * T_LEN + tt]);
    h[idx] = f2bf(s);
    outacc[idx] = 0.0f;
}

// ---------------------------------------------------------------------------
// Fused WaveNet layer: [dilated conv (K=384) || cond 1x1 (K=64)] -> gate ->
// res/skip 1x1 -> h_out / outacc.  One block = 64 timesteps of one batch row.
// 8 waves, bf16 WMMA 16x16x32.  Activation tiles are staged K-major so every
// B fragment is two aligned ds_load_b128s; ctx rides as rows 128..191 of the
// same tile (col offset dil) so the cond slabs share the conv code path.
// ---------------------------------------------------------------------------
template <bool LAST>
__global__ void __launch_bounds__(256, 1) k_layer(
    const float* __restrict__ w_in,    // (256,128,3)
    const float* __restrict__ w_cond,  // (1024,64)
    const float* __restrict__ b_in,    // (256)
    const float* __restrict__ b_cond,  // (1024)
    const float* __restrict__ w_rs,    // (M2,128)
    const float* __restrict__ b_rs,    // (M2)
    const unsigned short* __restrict__ h_in,
    const unsigned short* __restrict__ ctx,
    unsigned short* __restrict__ h_out,
    float* __restrict__ outacc,
    int dil, int condoff) {

    constexpr int M2   = LAST ? 128 : 256;
    constexpr int MTPW = LAST ? 1 : 2;   // M tiles per wave in GEMM2

    extern __shared__ char smem[];
    unsigned short* sx   = (unsigned short*)smem;            // [80 cols][192 rows] K-major
    unsigned short* sw   = sx + SXW * SXT;                   // 256 x 32 weight slab
    float*          sa   = (float*)(sw + 256 * 32);          // 256 x 64 f32 pre-activation
    unsigned short* sact = (unsigned short*)(sa + 256 * 64); // [64 cols][128 rows] K-major

    const int tid  = threadIdx.x;
    const int lane = tid & 31;
    const int wave = tid >> 5;
    const int t0   = blockIdx.x * NTILE;
    const int b    = blockIdx.y;
    const int wlim = NTILE + 2 * dil;

    __builtin_prefetch(w_rs, 0, 1);   // global_prefetch_b8: warm res/skip weights

    // stage h tile transposed (zero-padded halo): sx[col*SXT + ch]
    for (int i = tid; i < 128 * SXW; i += 256) {
        int ch = i / SXW, col = i % SXW;
        int g = t0 - dil + col;
        unsigned short v = 0;
        if (col < wlim && g >= 0 && g < T_LEN) v = h_in[(b * NC + ch) * T_LEN + g];
        sx[col * SXT + ch] = v;
    }
    // stage ctx as rows 128..191 at col = n + dil (matches center-tap shift)
    for (int i = tid; i < 64 * SXW; i += 256) {
        int cc = i / SXW, col = i % SXW;
        int n = col - dil;
        unsigned short v = 0;
        if (n >= 0 && n < NTILE) v = ctx[(b * 64 + cc) * T_LEN + t0 + n];
        sx[col * SXT + 128 + cc] = v;
    }
    __syncthreads();

    const __bf16* sxb = (const __bf16*)sx;
    const __bf16* swb = (const __bf16*)sw;
    const __bf16* sab = (const __bf16*)sact;

    const v8f vzero = {0.f, 0.f, 0.f, 0.f, 0.f, 0.f, 0.f, 0.f};
    v8f acc[8];
#pragma unroll
    for (int q = 0; q < 8; ++q) acc[q] = vzero;

    const int nn   = lane & 15;        // N / A-row within tile
    const int hi   = lane >> 4;        // lane half
    const int koff = hi ? 16 : 0;      // B fragment K base
    const int aoff = hi ? 8 : 0;       // C row offset

    // ---- GEMM1: a[256 x 64] = W[256 x 448] * X[448 x 64], 14 K-slabs of 32 ----
    for (int s = 0; s < 14; ++s) {
        for (int i = tid; i < 256 * 32; i += 256) {          // stage weight slab f32->bf16
            int m = i >> 5, j = i & 31;
            int kk = s * 32 + j;
            float w = (kk < 384) ? w_in[m * 384 + (kk & 127) * 3 + (kk >> 7)]
                                 : w_cond[(condoff + m) * 64 + (kk - 384)];
            sw[m * 32 + j] = f2bf(w);
        }
        __syncthreads();

        int rowbase, cshift;
        if (s < 12) { rowbase = (s & 3) << 5;         cshift = (s >> 2) * dil; }
        else        { rowbase = 128 + ((s - 12) << 5); cshift = dil; }
        const int rbk = rowbase + koff;

        v16bf af[2];                                          // A fragments (persistent)
#pragma unroll
        for (int mm = 0; mm < 2; ++mm) {
            int base = ((wave * 2 + mm) * 16 + nn) * 32 + koff;
#pragma unroll
            for (int h = 0; h < 8; ++h) { af[mm][h] = swb[base + h]; af[mm][8 + h] = swb[base + 8 + h]; }
        }
#pragma unroll
        for (int nt = 0; nt < 4; ++nt) {                      // one B fragment live at a time
            int cbase = (nt * 16 + nn + cshift) * SXT + rbk;  // 16 contiguous K -> 2x ds_load_b128
            v16bf bfv;
#pragma unroll
            for (int j = 0; j < 16; ++j) bfv[j] = sxb[cbase + j];
            acc[nt]     = __builtin_amdgcn_wmma_f32_16x16x32_bf16(false, af[0], false, bfv,
                                                                  (short)0, acc[nt], false, false);
            acc[4 + nt] = __builtin_amdgcn_wmma_f32_16x16x32_bf16(false, af[1], false, bfv,
                                                                  (short)0, acc[4 + nt], false, false);
        }
        __syncthreads();
    }

    // write pre-activation tile with conv + cond biases
#pragma unroll
    for (int mm = 0; mm < 2; ++mm) {
        int mt = wave * 2 + mm;
#pragma unroll
        for (int nt = 0; nt < 4; ++nt) {
            v8f c = acc[mm * 4 + nt];
#pragma unroll
            for (int r = 0; r < 8; ++r) {
                int m = mt * 16 + aoff + r;
                sa[m * NTILE + nt * 16 + nn] = c[r] + b_in[m] + b_cond[condoff + m];
            }
        }
    }
    __syncthreads();

    // gated activation: tanh(a[:128]) * sigmoid(a[128:]) -> K-major bf16 tile
    for (int i = tid; i < 128 * NTILE; i += 256) {
        int c = i / NTILE, n = i % NTILE;
        float tv = sa[c * NTILE + n];
        float sv = sa[(c + 128) * NTILE + n];
        sact[n * SAT + c] = f2bf(tanhf(tv) * (1.0f / (1.0f + expf(-sv))));
    }
    __syncthreads();

    // ---- GEMM2: rs[M2 x 64] = Wrs[M2 x 128] * acts[128 x 64], 4 K-slabs ----
#pragma unroll
    for (int q = 0; q < 8; ++q) acc[q] = vzero;
    for (int s = 0; s < 4; ++s) {
        for (int i = tid; i < M2 * 32; i += 256) {
            int m = i >> 5, j = i & 31;
            sw[m * 32 + j] = f2bf(w_rs[m * 128 + s * 32 + j]);
        }
        __syncthreads();
        const int k0 = s * 32 + koff;
        v16bf af[MTPW];
#pragma unroll
        for (int mm = 0; mm < MTPW; ++mm) {
            int base = ((wave * MTPW + mm) * 16 + nn) * 32 + koff;
#pragma unroll
            for (int h = 0; h < 8; ++h) { af[mm][h] = swb[base + h]; af[mm][8 + h] = swb[base + 8 + h]; }
        }
#pragma unroll
        for (int nt = 0; nt < 4; ++nt) {
            int cbase = (nt * 16 + nn) * SAT + k0;
            v16bf bfv;
#pragma unroll
            for (int j = 0; j < 16; ++j) bfv[j] = sab[cbase + j];
#pragma unroll
            for (int mm = 0; mm < MTPW; ++mm)
                acc[mm * 4 + nt] = __builtin_amdgcn_wmma_f32_16x16x32_bf16(
                    false, af[mm], false, bfv, (short)0, acc[mm * 4 + nt], false, false);
        }
        __syncthreads();
    }

    // epilogue: h_out = h_in + rs[:128], outacc += rs[128:] (last: outacc += rs)
#pragma unroll
    for (int mm = 0; mm < MTPW; ++mm) {
        int mt = wave * MTPW + mm;
#pragma unroll
        for (int nt = 0; nt < 4; ++nt) {
            v8f c = acc[mm * 4 + nt];
            int nloc = nt * 16 + nn;
            int gcol = t0 + nloc;
#pragma unroll
            for (int r = 0; r < 8; ++r) {
                int m = mt * 16 + aoff + r;
                float rs = c[r] + b_rs[m];
                if (!LAST) {
                    if (m < NC) {
                        float hnew = bf2f(sx[(dil + nloc) * SXT + m]) + rs;
                        h_out[(b * NC + m) * T_LEN + gcol] = f2bf(hnew);
                    } else {
                        outacc[(b * NC + (m - NC)) * T_LEN + gcol] += rs;
                    }
                } else {
                    outacc[(b * NC + m) * T_LEN + gcol] += rs;
                }
            }
        }
    }
}

// ---------------------------------------------------------------------------
// end 1x1 conv (2*n_half x 128) fused with affine coupling + log_s emission
// ---------------------------------------------------------------------------
__global__ void __launch_bounds__(256) k_end(const float* __restrict__ ew,
                                             const float* __restrict__ eb,
                                             const float* __restrict__ outacc,
                                             float* __restrict__ xw,
                                             float* __restrict__ lsout,
                                             int n_half, int ls_off) {
    int idx = blockIdx.x * blockDim.x + threadIdx.x;
    if (idx >= NBATCH * T_LEN) return;
    int b = idx / T_LEN, tt = idx % T_LEN;
    float o[8];
    int m2 = 2 * n_half;
    for (int j = 0; j < m2; ++j) {
        float s = eb[j];
        const float* wr = ew + j * NC;
        for (int c = 0; c < NC; ++c) s += wr[c] * outacc[(b * NC + c) * T_LEN + tt];
        o[j] = s;
    }
    for (int j = 0; j < n_half; ++j) {
        float ls = o[n_half + j];
        lsout[(b * 28 + ls_off + j) * T_LEN + tt] = ls;
        float* xp = &xw[(b * 8 + n_half + j) * T_LEN + tt];
        *xp = expf(ls) * (*xp) + o[j];
    }
}

// ---------------------------------------------------------------------------
// final: remaining 6 channels -> z channels 2..7
// ---------------------------------------------------------------------------
__global__ void __launch_bounds__(256) k_final(const float* __restrict__ xw,
                                               float* __restrict__ z) {
    int idx = blockIdx.x * blockDim.x + threadIdx.x;
    if (idx >= NBATCH * T_LEN) return;
    int b = idx / T_LEN, tt = idx % T_LEN;
    for (int r = 0; r < 6; ++r)
        z[(b * 8 + 2 + r) * T_LEN + tt] = xw[(b * 8 + r) * T_LEN + tt];
}

// ---------------------------------------------------------------------------
extern "C" void kernel_launch(void* const* d_in, const int* in_sizes, int n_in,
                              void* d_out, int out_size, void* d_ws, size_t ws_size,
                              hipStream_t stream) {
    const float* forecast = (const float*)d_in[0];
    const float* context  = (const float*)d_in[1];
    // per-flow param layout (dict insertion order), 23 leaves per flow:
    // 0:ic_w 1:start_w 2:start_b 3:cond_w 4:cond_b 5-8:in_w 9-12:in_b
    // 13-16:rs_w 17-20:rs_b 21:end_w 22:end_b
    auto FP = [&](int k, int idx) { return (const float*)d_in[2 + k * 23 + idx]; };

    char* ws = (char*)d_ws;
    size_t off = 0;
    auto alloc = [&](size_t bytes) {
        char* p = ws + off;
        off += (bytes + 255) & ~(size_t)255;
        return p;
    };
    unsigned short* ctx_bf = (unsigned short*)alloc((size_t)NBATCH * 64 * T_LEN * 2);
    float*          xw     = (float*)         alloc((size_t)NBATCH * 8  * T_LEN * 4);
    unsigned short* x0bf   = (unsigned short*)alloc((size_t)NBATCH * 4  * T_LEN * 2);
    unsigned short* h0     = (unsigned short*)alloc((size_t)NBATCH * NC * T_LEN * 2);
    unsigned short* h1     = (unsigned short*)alloc((size_t)NBATCH * NC * T_LEN * 2);
    float*          outacc = (float*)         alloc((size_t)NBATCH * NC * T_LEN * 4);

    float* z_out  = (float*)d_out;                   // (B, 8, T)
    float* ls_out = z_out + NBATCH * 8 * T_LEN;      // (B, 28, T)
    float* ld_out = ls_out + NBATCH * 28 * T_LEN;    // (8,)

    const int dils[4] = {1, 2, 4, 8};
    const size_t SMEM = SXW * SXT * 2 + 256 * 32 * 2 + 256 * 64 * 4 + 64 * SAT * 2;

    k_prep<<<(NBATCH * 64 * T_LEN + 255) / 256, 256, 0, stream>>>(context, ctx_bf, forecast, xw);
    k_logdet<<<1, 32, 0, stream>>>(FP(0,0), FP(1,0), FP(2,0), FP(3,0),
                                   FP(4,0), FP(5,0), FP(6,0), FP(7,0), ld_out);

    int ls_off = 0;
    for (int k = 0; k < 8; ++k) {
        int n_half = (k < 4) ? 4 : 3;
        int nrem   = 2 * n_half;
        if (k == 4)
            k_early<<<(NBATCH * T_LEN + 255) / 256, 256, 0, stream>>>(xw, z_out);
        k_invconv<<<(NBATCH * T_LEN + 255) / 256, 256, 0, stream>>>(FP(k,0), xw, x0bf, nrem, n_half);
        k_start<<<(NBATCH * NC * T_LEN + 255) / 256, 256, 0, stream>>>(
            FP(k,1), FP(k,2), x0bf, h0, outacc, n_half);

        const unsigned short* hin = h0;
        unsigned short* hout = h1;
        for (int i = 0; i < 4; ++i) {
            if (i == 3)
                k_layer<true><<<dim3(T_LEN / NTILE, NBATCH), 256, SMEM, stream>>>(
                    FP(k, 5 + i), FP(k, 3), FP(k, 9 + i), FP(k, 4), FP(k, 13 + i), FP(k, 17 + i),
                    hin, ctx_bf, hout, outacc, dils[i], i * 256);
            else
                k_layer<false><<<dim3(T_LEN / NTILE, NBATCH), 256, SMEM, stream>>>(
                    FP(k, 5 + i), FP(k, 3), FP(k, 9 + i), FP(k, 4), FP(k, 13 + i), FP(k, 17 + i),
                    hin, ctx_bf, hout, outacc, dils[i], i * 256);
            const unsigned short* t = hin; hin = hout; hout = (unsigned short*)t;
        }
        k_end<<<(NBATCH * T_LEN + 255) / 256, 256, 0, stream>>>(
            FP(k, 21), FP(k, 22), outacc, xw, ls_out, n_half, ls_off);
        ls_off += n_half;
    }
    k_final<<<(NBATCH * T_LEN + 255) / 256, 256, 0, stream>>>(xw, z_out);

    (void)in_sizes; (void)n_in; (void)out_size; (void)ws_size;
}
